// BinConvBn_9491877724616
// MI455X (gfx1250) — 8-bit matrix operand order repaired and checked by reading, compiled, not run on hardware
//
#include <hip/hip_runtime.h>
#include <stdint.h>

typedef __attribute__((ext_vector_type(8))) int v8i;
typedef unsigned long long ull;

#define IN_CH   256
#define OUT_CH  256
#define Hh      56
#define Ww      56
#define NIMG    32
#define HW      (Hh*Ww)
#define PIX     (NIMG*HW)
#define KSTEPS  36
#define NTILES  (OUT_CH/16)
#define WQ_BYTES ((size_t)NTILES * KSTEPS * 1024)

__global__ void __launch_bounds__(256)
binarize_x_kernel(const float* __restrict__ x, int8_t* __restrict__ xq) {
    int tid = blockIdx.x * blockDim.x + threadIdx.x;
    int pix = tid % PIX;
    int ic4 = tid / PIX;
    int img = pix / HW;
    int hw  = pix - img * HW;

    const float* xp = x + ((size_t)img * IN_CH + ic4 * 4) * HW + hw;
    char4 s;
    float v0 = xp[0 * HW], v1 = xp[1 * HW], v2 = xp[2 * HW], v3 = xp[3 * HW];
    s.x = (int8_t)((v0 > 0.f) - (v0 < 0.f));
    s.y = (int8_t)((v1 > 0.f) - (v1 < 0.f));
    s.z = (int8_t)((v2 > 0.f) - (v2 < 0.f));
    s.w = (int8_t)((v3 > 0.f) - (v3 < 0.f));
    *(char4*)(xq + (size_t)pix * IN_CH + ic4 * 4) = s;
}

__global__ void __launch_bounds__(256)
pack_w_kernel(const float* __restrict__ wt, int8_t* __restrict__ wq,
              int8_t* __restrict__ zp) {
    int idx = blockIdx.x * blockDim.x + threadIdx.x;
    if (idx < IN_CH) zp[idx] = 0;

    int byteInLane = idx & 31;
    int lane  = (idx >> 5) & 31;
    int step  = (idx >> 10) % KSTEPS;
    int tileN = idx / (KSTEPS * 1024);
    int half  = lane >> 4;
    int n     = lane & 15;
    int v     = byteInLane >> 2;
    int b     = byteInLane & 3;
    int K     = (v >> 1) * 16 + half * 8 + (v & 1) * 4 + b;
    int c     = step & 3;
    int kk    = step >> 2;
    int kh    = kk / 3, kw = kk - kh * 3;
    int ic    = c * 64 + K;
    int oc    = tileN * 16 + n;
    float wv  = wt[(((size_t)oc * IN_CH + ic) * 3 + kh) * 3 + kw];
    wq[idx]   = (int8_t)((wv > 0.f) - (wv < 0.f));
}

__global__ void __launch_bounds__(256)
binconv_wmma_kernel(const int8_t* __restrict__ xq, const int8_t* __restrict__ wq,
                    const int8_t* __restrict__ zp,
                    const float* __restrict__ scale, float* __restrict__ out) {
    const int lane   = threadIdx.x & 31;
    const int wave   = threadIdx.x >> 5;
    const int tileM  = blockIdx.x * 8 + wave;
    const int tileN0 = blockIdx.y * 4;
    const int half   = lane >> 4;
    const int m      = lane & 15;

    const int p   = tileM * 16 + m;
    const int img = p / HW;
    const int hw  = p - img * HW;
    const int h   = hw / Ww;
    const int w   = hw - h * Ww;

    const int8_t* __restrict__ xbase = xq + (size_t)img * HW * IN_CH;
    const int8_t* __restrict__ bbase =
        wq + (size_t)tileN0 * KSTEPS * 1024 + (size_t)lane * 32;

    v8i acc0 = {0,0,0,0,0,0,0,0};
    v8i acc1 = {0,0,0,0,0,0,0,0};
    v8i acc2 = {0,0,0,0,0,0,0,0};
    v8i acc3 = {0,0,0,0,0,0,0,0};

    #pragma unroll
    for (int kk = 0; kk < 9; ++kk) {
        const int kh = kk / 3, kw = kk - kh * 3;
        const int ih = h + kh - 1, iw = w + kw - 1;
        const bool valid = ((unsigned)ih < (unsigned)Hh) & ((unsigned)iw < (unsigned)Ww);
        const int8_t* arow =
            (valid ? (xbase + (size_t)(ih * Ww + iw) * IN_CH) : zp) + half * 8;

        #pragma unroll
        for (int c = 0; c < 4; ++c) {
            const int8_t* ap = arow + c * 64;
            ull a0 = *(const ull*)(ap +  0);
            ull a1 = *(const ull*)(ap + 16);
            ull a2 = *(const ull*)(ap + 32);
            ull a3 = *(const ull*)(ap + 48);
            v8i A;
            A[0] = (int)a0; A[1] = (int)(a0 >> 32);
            A[2] = (int)a1; A[3] = (int)(a1 >> 32);
            A[4] = (int)a2; A[5] = (int)(a2 >> 32);
            A[6] = (int)a3; A[7] = (int)(a3 >> 32);

            const size_t boff = (size_t)(kk * 4 + c) * 1024;
            v8i B0 = *(const v8i*)(bbase + boff);
            v8i B1 = *(const v8i*)(bbase + boff + 1 * KSTEPS * 1024);
            v8i B2 = *(const v8i*)(bbase + boff + 2 * KSTEPS * 1024);
            v8i B3 = *(const v8i*)(bbase + boff + 3 * KSTEPS * 1024);

            acc0 = __builtin_amdgcn_wmma_i32_16x16x64_iu8(true, A, true, B0, acc0, false, false);
            acc1 = __builtin_amdgcn_wmma_i32_16x16x64_iu8(true, A, true, B1, acc1, false, false);
            acc2 = __builtin_amdgcn_wmma_i32_16x16x64_iu8(true, A, true, B2, acc2, false, false);
            acc3 = __builtin_amdgcn_wmma_i32_16x16x64_iu8(true, A, true, B3, acc3, false, false);
        }
    }

    const int p0   = tileM * 16 + half * 8;
    const int img0 = p0 / HW;
    const int hw0  = p0 - img0 * HW;
    #pragma unroll
    for (int t = 0; t < 4; ++t) {
        const v8i& a = (t == 0) ? acc0 : (t == 1) ? acc1 : (t == 2) ? acc2 : acc3;
        const int oc  = (tileN0 + t) * 16 + (lane & 15);
        const float s = scale[oc];
        float* __restrict__ ob = out + ((size_t)img0 * OUT_CH + oc) * HW + hw0;
        #pragma unroll
        for (int r = 0; r < 8; ++r)
            ob[r] = (float)a[r] * s;
    }
}

extern "C" void kernel_launch(void* const* d_in, const int* in_sizes, int n_in,
                              void* d_out, int out_size, void* d_ws, size_t ws_size,
                              hipStream_t stream) {
    const float* x     = (const float*)d_in[0];
    const float* wt    = (const float*)d_in[1];
    const float* scale = (const float*)d_in[2];
    float* out         = (float*)d_out;

    int8_t* xq = (int8_t*)d_ws;
    int8_t* wq = xq + (size_t)PIX * IN_CH;
    int8_t* zp = wq + WQ_BYTES;

    {
        int total = (IN_CH / 4) * PIX;
        binarize_x_kernel<<<total / 256, 256, 0, stream>>>(x, xq);
    }
    {
        int total = NTILES * KSTEPS * 1024;
        pack_w_kernel<<<total / 256, 256, 0, stream>>>(wt, wq, zp);
    }
    {
        dim3 grid(PIX / 16 / 8, NTILES / 4);
        binconv_wmma_kernel<<<grid, 256, 0, stream>>>(xq, wq, zp, scale, out);
    }
}
